// GCN3layer_57535381897260
// MI455X (gfx1250) — matrix operand order repaired and checked by reading, not compiled, not run on hardware
//
#include <hip/hip_runtime.h>
#include <hip/hip_bf16.h>
#include <math.h>

typedef __attribute__((ext_vector_type(16))) __bf16 v16bf;
typedef __attribute__((ext_vector_type(8)))  float  v8f;

#define GCN_NODES 50000
#define GCN_NFEAT 500
#define GCN_NHID  128
#define GCN_NCLS  40

// ------------------------- degree / normalization -------------------------
__global__ void k_init_deg(float* __restrict__ deg, int n) {
  int i = blockIdx.x * blockDim.x + threadIdx.x;
  if (i < n) deg[i] = 1.0f;  // self-loop contribution
}

__global__ void k_deg_accum(const int* __restrict__ dst, float* __restrict__ deg, int e) {
  int i = blockIdx.x * blockDim.x + threadIdx.x;
  if (i < e) atomicAdd(&deg[dst[i]], 1.0f);
}

__global__ void k_rsqrt_inplace(float* __restrict__ d, int n) {
  int i = blockIdx.x * blockDim.x + threadIdx.x;
  if (i < n) d[i] = rsqrtf(d[i]);  // deg >= 1 always (self-loop)
}

// ------------------------- WMMA bf16 GEMM: H = X @ W -------------------------
// X: [nrows, K] f32 row-major, W: [K, NOUT] f32 row-major, H: [nrows, NOUT] f32.
// 256 threads = 8 waves; each wave computes 16 rows x NOUT cols.
// W is staged to LDS as bf16 in K-chunks of 128, stored FRAGMENT-MAJOR:
// for (ktile, lane-half g, column n) the 16 K-elements a lane consumes are
// contiguous -> each B fragment is one aligned 32B LDS read.
template <int K, int NOUT>
__global__ void __launch_bounds__(256)
k_gemm_bf16(const float* __restrict__ X, const float* __restrict__ W,
            float* __restrict__ H, int nrows) {
  constexpr int KCHUNK = 128;
  constexpr int NPAD   = ((NOUT + 15) / 16) * 16;
  constexpr int NT     = NPAD / 16;

  __shared__ __align__(32) __bf16 sW[KCHUNK * NPAD];  // fragment-major

  const int tid     = threadIdx.x;
  const int lane    = tid & 31;
  const int wave    = tid >> 5;
  const int g       = lane >> 4;   // lane-half (ISA 16-bit fragment layouts)
  const int r16     = lane & 15;
  const int rowTile = blockIdx.x * 8 + wave;
  const int mtiles  = nrows >> 4;            // nrows is a multiple of 16
  const bool active = rowTile < mtiles;      // wave-uniform -> EXEC all-1 at WMMA
  const long rowBase = (long)rowTile * 16;

  v8f acc[NT];
#pragma unroll
  for (int t = 0; t < NT; ++t) { v8f z = {}; acc[t] = z; }

  const float* xrow = X + (rowBase + r16) * (long)K;

  for (int kk = 0; kk < K; kk += KCHUNK) {
    __syncthreads();
    // Cooperative staging of W chunk -> LDS (f32 -> bf16, zero-padded),
    // written in fragment-major layout:
    //   dest = ((ktile*2 + g) * NPAD + n) * 16 + e   with k = ktile*32 + g*8 + (e&7) + (e>>3)*16
    for (int idx = tid; idx < KCHUNK * NPAD; idx += 256) {
      int kl = idx / NPAD;            // local k within chunk
      int n  = idx - kl * NPAD;
      int kg = kk + kl;
      float v = (kg < K && n < NOUT) ? W[(long)kg * NOUT + n] : 0.0f;
      int frag = (kl >> 5) * 2 + ((kl >> 3) & 1);
      sW[(frag * NPAD + n) * 16 + ((kl & 7) | ((kl >> 1) & 8))] = (__bf16)v;
    }
    __syncthreads();

    if (active) {
      // Prefetch next K-chunk of this row while computing on current one.
      if (kk + KCHUNK < K) __builtin_prefetch(xrow + kk + KCHUNK, 0, 0);

      int kend  = K - kk; if (kend > KCHUNK) kend = KCHUNK;
      int kfull = kend >> 5;           // full (unchecked) 32-wide K-steps

      for (int ks = 0; ks < kfull; ++ks) {
        // A fragment 16x32 bf16: lane-half g holds K runs [g*8,+8) and [16+g*8,+8).
        // Rows of X are 16B-aligned (K*4 % 16 == 0), runs are 32B-aligned.
        v16bf a;
        int kb = kk + ks * 32 + g * 8;
        const float4* pa = (const float4*)(xrow + kb);
        const float4* pb = (const float4*)(xrow + kb + 16);
        float4 q0 = pa[0], q1 = pa[1];
        float4 q2 = pb[0], q3 = pb[1];
        a[0]  = (__bf16)q0.x; a[1]  = (__bf16)q0.y; a[2]  = (__bf16)q0.z; a[3]  = (__bf16)q0.w;
        a[4]  = (__bf16)q1.x; a[5]  = (__bf16)q1.y; a[6]  = (__bf16)q1.z; a[7]  = (__bf16)q1.w;
        a[8]  = (__bf16)q2.x; a[9]  = (__bf16)q2.y; a[10] = (__bf16)q2.z; a[11] = (__bf16)q2.w;
        a[12] = (__bf16)q3.x; a[13] = (__bf16)q3.y; a[14] = (__bf16)q3.z; a[15] = (__bf16)q3.w;

        int fragBase = (ks * 2 + g) * NPAD;
#pragma unroll
        for (int ct = 0; ct < NT; ++ct) {
          int n = ct * 16 + r16;
          v16bf b = *(const v16bf*)(sW + (fragBase + n) * 16);  // 32B aligned
          acc[ct] = __builtin_amdgcn_wmma_f32_16x16x32_bf16(
              false, a, false, b, (short)0, acc[ct], false, false);
        }
      }

      if (kend & 31) {  // single branchless tail step (only K=500 last chunk)
        int ks = kfull;
        v16bf a;
        int kb = kk + ks * 32 + g * 8;
#pragma unroll
        for (int e2 = 0; e2 < 8; ++e2) {
          int k0 = kb + e2;
          int k1 = kb + 16 + e2;
          // clamp address, select value -> no EXEC-mask branches
          float v0 = xrow[k0 < K ? k0 : K - 1];
          float v1 = xrow[k1 < K ? k1 : K - 1];
          a[e2]     = (k0 < K) ? (__bf16)v0 : (__bf16)0.0f;
          a[e2 + 8] = (k1 < K) ? (__bf16)v1 : (__bf16)0.0f;
        }
        int fragBase = (ks * 2 + g) * NPAD;
#pragma unroll
        for (int ct = 0; ct < NT; ++ct) {
          int n = ct * 16 + r16;
          v16bf b = *(const v16bf*)(sW + (fragBase + n) * 16);  // zero-padded k>=K
          acc[ct] = __builtin_amdgcn_wmma_f32_16x16x32_bf16(
              false, a, false, b, (short)0, acc[ct], false, false);
        }
      }
    }
  }

  if (active) {
#pragma unroll
    for (int ct = 0; ct < NT; ++ct) {
      int n = ct * 16 + r16;
      if (n < NOUT) {
#pragma unroll
        for (int i2 = 0; i2 < 8; ++i2) {
          long m = rowBase + g * 8 + i2;   // C/D layout: elem i -> row (g*8+i)
          H[m * NOUT + n] = acc[ct][i2];
        }
      }
    }
  }
}

// ------------------------- aggregation -------------------------
// OUT[i,f..f+3] = bias[f..f+3] + H[i,f..f+3]*dinv[i]^2  (bias + self-loop term)
// F % 4 == 0 for both 128 and 40, so each float4 stays within one row.
template <int F>
__global__ void k_agg_init(const float* __restrict__ H, const float* __restrict__ bias,
                           const float* __restrict__ dinv, float* __restrict__ OUT, int n) {
  int q = blockIdx.x * blockDim.x + threadIdx.x;
  if (q >= n * (F / 4)) return;
  int idx = q * 4;
  int i = idx / F;
  int f = idx - i * F;
  float di = dinv[i];
  float s = di * di;
  float4 h  = *(const float4*)(H + idx);
  float4 bb = *(const float4*)(bias + f);
  float4 o;
  o.x = bb.x + h.x * s;
  o.y = bb.y + h.y * s;
  o.z = bb.z + h.z * s;
  o.w = bb.w + h.w * s;
  *(float4*)(OUT + idx) = o;
}

// One wave32 per edge: OUT[dst] += H[src] * dinv[src]*dinv[dst]
template <int F>
__global__ void k_agg_edges(const int* __restrict__ src, const int* __restrict__ dst,
                            const float* __restrict__ H, const float* __restrict__ dinv,
                            float* __restrict__ OUT, int e) {
  int w    = (blockIdx.x * blockDim.x + threadIdx.x) >> 5;  // global wave id
  int lane = threadIdx.x & 31;
  if (w >= e) return;                                        // wave-uniform
  int s = src[w];
  int d = dst[w];
  float nrm = dinv[s] * dinv[d];
  const float* hs = H + (long)s * F;
  float* od = OUT + (long)d * F;
#pragma unroll
  for (int f = lane; f < F; f += 32) atomicAdd(&od[f], hs[f] * nrm);
}

__global__ void k_relu4(float* __restrict__ a, int nquads) {
  int q = blockIdx.x * blockDim.x + threadIdx.x;
  if (q >= nquads) return;
  float4 v = *(const float4*)(a + q * 4);
  v.x = fmaxf(v.x, 0.0f);
  v.y = fmaxf(v.y, 0.0f);
  v.z = fmaxf(v.z, 0.0f);
  v.w = fmaxf(v.w, 0.0f);
  *(float4*)(a + q * 4) = v;
}

// ------------------------- log-softmax over NCLS -------------------------
__global__ void k_log_softmax(const float* __restrict__ A, float* __restrict__ OUT, int n) {
  int i = blockIdx.x * blockDim.x + threadIdx.x;
  if (i >= n) return;
  const float* r = A + (long)i * GCN_NCLS;
  float m = -INFINITY;
#pragma unroll
  for (int c = 0; c < GCN_NCLS; ++c) m = fmaxf(m, r[c]);
  float s = 0.0f;
#pragma unroll
  for (int c = 0; c < GCN_NCLS; ++c) s += expf(r[c] - m);
  float ls = logf(s);
  float* o = OUT + (long)i * GCN_NCLS;
#pragma unroll
  for (int c = 0; c < GCN_NCLS; ++c) o[c] = r[c] - m - ls;
}

// ------------------------- driver -------------------------
extern "C" void kernel_launch(void* const* d_in, const int* in_sizes, int n_in,
                              void* d_out, int out_size, void* d_ws, size_t ws_size,
                              hipStream_t stream) {
  (void)n_in; (void)out_size; (void)ws_size;
  const float* x  = (const float*)d_in[0];
  const int*   ei = (const int*)d_in[1];   // [2, E] int32 (JAX default x64-off)
  const float* W1 = (const float*)d_in[2];
  const float* b1 = (const float*)d_in[3];
  const float* W2 = (const float*)d_in[4];
  const float* b2 = (const float*)d_in[5];
  const float* W3 = (const float*)d_in[6];
  const float* b3 = (const float*)d_in[7];
  float* out = (float*)d_out;

  const int n = GCN_NODES;
  const int e = in_sizes[1] / 2;
  const int* srcp = ei;
  const int* dstp = ei + e;

  // workspace layout: dinv | bufH | bufA
  char* ws = (char*)d_ws;
  float* dinv = (float*)ws;
  size_t off  = (((size_t)n * 4) + 255) & ~(size_t)255;
  float* bufH = (float*)(ws + off);
  float* bufA = bufH + (size_t)n * GCN_NHID;

  const int tpb = 256;
  const int gN     = (n + tpb - 1) / tpb;
  const int gE     = (e + tpb - 1) / tpb;
  const int gEw    = (e + 7) / 8;                       // 8 waves (edges) per block
  const int gHid4  = (n * (GCN_NHID / 4) + tpb - 1) / tpb;
  const int gCls4  = (n * (GCN_NCLS / 4) + tpb - 1) / tpb;
  const int gGemm  = ((n / 16) + 7) / 8;                // 8 row-tiles per block

  // D^{-1/2}
  k_init_deg<<<gN, tpb, 0, stream>>>(dinv, n);
  k_deg_accum<<<gE, tpb, 0, stream>>>(dstp, dinv, e);
  k_rsqrt_inplace<<<gN, tpb, 0, stream>>>(dinv, n);

  // layer 1: [N,500] @ [500,128]
  k_gemm_bf16<GCN_NFEAT, GCN_NHID><<<gGemm, tpb, 0, stream>>>(x, W1, bufH, n);
  k_agg_init<GCN_NHID><<<gHid4, tpb, 0, stream>>>(bufH, b1, dinv, bufA, n);
  k_agg_edges<GCN_NHID><<<gEw, tpb, 0, stream>>>(srcp, dstp, bufH, dinv, bufA, e);
  k_relu4<<<gHid4, tpb, 0, stream>>>(bufA, n * (GCN_NHID / 4));

  // layer 2: [N,128] @ [128,128]
  k_gemm_bf16<GCN_NHID, GCN_NHID><<<gGemm, tpb, 0, stream>>>(bufA, W2, bufH, n);
  k_agg_init<GCN_NHID><<<gHid4, tpb, 0, stream>>>(bufH, b2, dinv, bufA, n);
  k_agg_edges<GCN_NHID><<<gEw, tpb, 0, stream>>>(srcp, dstp, bufH, dinv, bufA, e);
  k_relu4<<<gHid4, tpb, 0, stream>>>(bufA, n * (GCN_NHID / 4));

  // layer 3: [N,128] @ [128,40]
  k_gemm_bf16<GCN_NHID, GCN_NCLS><<<gGemm, tpb, 0, stream>>>(bufA, W3, bufH, n);
  k_agg_init<GCN_NCLS><<<gCls4, tpb, 0, stream>>>(bufH, b3, dinv, bufA, n);
  k_agg_edges<GCN_NCLS><<<gEw, tpb, 0, stream>>>(srcp, dstp, bufH, dinv, bufA, e);

  k_log_softmax<<<gN, tpb, 0, stream>>>(bufA, out, n);
}
